// CausalSelfAttention_66675072303434
// MI455X (gfx1250) — matrix operand order repaired and checked by reading, not compiled, not run on hardware
//
#include <hip/hip_runtime.h>

typedef __bf16 bf16;
typedef __attribute__((ext_vector_type(16))) __bf16 v16bf;
typedef __attribute__((ext_vector_type(8)))  __bf16 v8bf;
typedef __attribute__((ext_vector_type(4)))  __bf16 v4bf;
typedef __attribute__((ext_vector_type(8)))  float  v8f;

#define D_MODEL 1024
#define NH 16
#define DH 64
#define BATCH 4
#define SEQ 2048
#define M_TOTAL (BATCH*SEQ)   // 8192

// Async global->LDS 16B copy: per-lane LDS byte offset + 64-bit global address.
// GV mode (saddr = off). Tracked by ASYNCcnt.
__device__ __forceinline__ void async_copy16(unsigned lds_off, const void* gptr) {
    asm volatile("global_load_async_to_lds_b128 %0, %1, off"
                 :: "v"(lds_off), "v"(gptr) : "memory");
}
__device__ __forceinline__ void wait_async0() {
    asm volatile("s_wait_asynccnt 0" ::: "memory");
}
__device__ __forceinline__ unsigned lds_off_of(const void* p) {
    // LDS aperture: flat addr[31:0] == LDS byte offset
    return (unsigned)(unsigned long long)p;
}

// ---------------- fp32 -> bf16 convert (4-wide) ----------------
__global__ void cvt_f32_bf16(const float* __restrict__ s, bf16* __restrict__ d, int n) {
    int i = (blockIdx.x * blockDim.x + threadIdx.x) * 4;
    if (i < n) {
        float4 v = *(const float4*)(s + i);
        v4bf o;
        o[0] = (bf16)v.x; o[1] = (bf16)v.y; o[2] = (bf16)v.z; o[3] = (bf16)v.w;
        *(v4bf*)(d + i) = o;
    }
}

// ---------------- bf16 GEMM: C[M,N] = A[M,K] * W[N,K]^T ----------------
// Block tile 128x128, 256 threads = 8 waves (2x4), wave tile 64x32.
// Double-buffered LDS fed by async global->LDS copies; one barrier per stage.
template<bool OUT_BF16>
__global__ __launch_bounds__(256)
void gemm_bf16(const bf16* __restrict__ A, const bf16* __restrict__ W,
               void* __restrict__ Cout, int M, int N, int K) {
    constexpr int LDT = 72;              // padded LDS row stride (9 * 16B)
    __shared__ bf16 Asm[2][128 * LDT];
    __shared__ bf16 Bsm[2][128 * LDT];

    const int tid  = threadIdx.x;
    const int lane = tid & 31;
    const int wid  = tid >> 5;
    const int kh   = lane >> 4;          // which half of the wave
    const int l16  = lane & 15;
    const int waveM = (wid >> 2) * 64;
    const int waveN = (wid & 3) * 32;
    const int rowBase = blockIdx.y * 128;
    const int colBase = blockIdx.x * 128;

    v8f acc[4][2];
    #pragma unroll
    for (int i = 0; i < 4; i++)
        #pragma unroll
        for (int j = 0; j < 2; j++)
            acc[i][j] = (v8f){0.f,0.f,0.f,0.f,0.f,0.f,0.f,0.f};

    // Issue async copies of one 128x64 A tile + W tile into LDS buffer `buf`.
    auto async_stage = [&](int k0, int buf) {
        #pragma unroll
        for (int i = 0; i < 4; i++) {
            int c = tid + i * 256;
            int r = c >> 3, ch = c & 7;
            async_copy16(lds_off_of(&Asm[buf][r * LDT + ch * 8]),
                         &A[(size_t)(rowBase + r) * K + k0 + ch * 8]);
            async_copy16(lds_off_of(&Bsm[buf][r * LDT + ch * 8]),
                         &W[(size_t)(colBase + r) * K + k0 + ch * 8]);
        }
    };

    async_stage(0, 0);
    wait_async0();
    __syncthreads();

    const int niter = K >> 6;            // K/64 stages
    for (int it = 0; it < niter; it++) {
        const int buf = it & 1;
        if (it + 1 < niter) async_stage((it + 1) * 64, buf ^ 1);  // overlaps compute
        if (it + 2 < niter) {            // warm L2 for the stage after next
            int c = tid;
            int r = c >> 3, ch = c & 7;
            __builtin_prefetch(&A[(size_t)(rowBase + r) * K + (it + 2) * 64 + ch * 8], 0, 1);
            __builtin_prefetch(&W[(size_t)(colBase + r) * K + (it + 2) * 64 + ch * 8], 0, 1);
        }

        #pragma unroll
        for (int kk = 0; kk < 64; kk += 32) {
            v16bf af[4], bfm[2];
            #pragma unroll
            for (int mt = 0; mt < 4; mt++) {
                const bf16* p = &Asm[buf][(waveM + mt * 16 + l16) * LDT + kk + 8 * kh];
                v8bf lo = *(const v8bf*)p;
                v8bf hi = *(const v8bf*)(p + 16);
                #pragma unroll
                for (int e = 0; e < 8; e++) { af[mt][e] = lo[e]; af[mt][e + 8] = hi[e]; }
            }
            #pragma unroll
            for (int nt = 0; nt < 2; nt++) {
                const bf16* p = &Bsm[buf][(waveN + nt * 16 + l16) * LDT + kk + 8 * kh];
                v8bf lo = *(const v8bf*)p;
                v8bf hi = *(const v8bf*)(p + 16);
                #pragma unroll
                for (int e = 0; e < 8; e++) { bfm[nt][e] = lo[e]; bfm[nt][e + 8] = hi[e]; }
            }
            #pragma unroll
            for (int mt = 0; mt < 4; mt++)
                #pragma unroll
                for (int nt = 0; nt < 2; nt++)
                    acc[mt][nt] = __builtin_amdgcn_wmma_f32_16x16x32_bf16(
                        false, af[mt], false, bfm[nt], (short)0, acc[mt][nt], false, false);
        }

        wait_async0();                   // next tile resident in other buffer
        __syncthreads();
    }

    // Epilogue: C layout VGPR r -> row r+8*kh, col l16 (coalesced across lanes)
    #pragma unroll
    for (int mt = 0; mt < 4; mt++)
        #pragma unroll
        for (int nt = 0; nt < 2; nt++)
            #pragma unroll
            for (int r = 0; r < 8; r++) {
                size_t grow = rowBase + waveM + mt * 16 + r + 8 * kh;
                size_t gcol = colBase + waveN + nt * 16 + l16;
                if (OUT_BF16)
                    ((bf16*)Cout)[grow * N + gcol] = (bf16)acc[mt][nt][r];
                else
                    ((float*)Cout)[grow * N + gcol] = acc[mt][nt][r];
            }
}

// ---------------- Flash attention (causal, streaming softmax) ----------------
// Grid: (SEQ/128, NH, BATCH). 256 threads = 8 waves; each wave owns 16 query rows.
// K staged by async global->LDS copies; V staged via registers (needs transpose).
// Double-buffered, one barrier per 64-key block; causally-dead blocks skip compute.
__global__ __launch_bounds__(256)
void flash_attn(const bf16* __restrict__ qkv, bf16* __restrict__ attout) {
    constexpr int LDT = 72;
    __shared__ bf16 Ksm[2][64 * LDT];     // K block, [key][dim]
    __shared__ bf16 Vt [2][64 * LDT];     // V block transposed, [dim][key]
    __shared__ bf16 Psm[8 * 16 * LDT];    // per-wave P tiles, [row][key]

    const int tid  = threadIdx.x;
    const int lane = tid & 31;
    const int wid  = tid >> 5;
    const int kh   = lane >> 4;
    const int l16  = lane & 15;
    const int qb = blockIdx.x, h = blockIdx.y, b = blockIdx.z;

    const size_t bt = (size_t)b * SEQ;
    const int qrow0 = qb * 128 + wid * 16;

    // Q fragments for this wave's 16 rows (2 k-steps over Dh=64)
    v16bf qf[2];
    {
        const bf16* qp = qkv + (bt + qrow0 + l16) * (3 * D_MODEL) + h * DH;
        #pragma unroll
        for (int s = 0; s < 2; s++) {
            v8bf lo = *(const v8bf*)(qp + 32 * s + 8 * kh);
            v8bf hi = *(const v8bf*)(qp + 32 * s + 16 + 8 * kh);
            #pragma unroll
            for (int e = 0; e < 8; e++) { qf[s][e] = lo[e]; qf[s][e + 8] = hi[e]; }
        }
    }

    float m_i[8], l_i[8];
    v8f o[4];
    #pragma unroll
    for (int r = 0; r < 8; r++) { m_i[r] = -1e30f; l_i[r] = 0.f; }
    #pragma unroll
    for (int nt = 0; nt < 4; nt++) o[nt] = (v8f){0.f,0.f,0.f,0.f,0.f,0.f,0.f,0.f};

    bf16* Pw = &Psm[wid * 16 * LDT];

    // K staging: async copies straight into LDS (contiguous layout).
    auto async_k = [&](int kbase, int buf) {
        #pragma unroll
        for (int i = 0; i < 2; i++) {
            int c = tid + i * 256;
            int r = c >> 3, ch = c & 7;
            async_copy16(lds_off_of(&Ksm[buf][r * LDT + ch * 8]),
                         qkv + (bt + kbase + r) * (3 * D_MODEL) + D_MODEL + h * DH + ch * 8);
        }
    };
    // V staging: registers, stored transposed.
    v8bf vreg[2];
    auto load_v = [&](int kbase) {
        #pragma unroll
        for (int i = 0; i < 2; i++) {
            int c = tid + i * 256;
            int r = c >> 3, ch = c & 7;
            vreg[i] = *(const v8bf*)(qkv + (bt + kbase + r) * (3 * D_MODEL)
                                     + 2 * D_MODEL + h * DH + ch * 8);
        }
    };
    auto store_v = [&](int buf) {
        #pragma unroll
        for (int i = 0; i < 2; i++) {
            int c = tid + i * 256;
            int r = c >> 3, ch = c & 7;
            #pragma unroll
            for (int e = 0; e < 8; e++) Vt[buf][(ch * 8 + e) * LDT + r] = vreg[i][e];
        }
    };

    const int nkb = 2 * qb + 2;          // key blocks covering keys <= qb*128+127
    async_k(0, 0);
    load_v(0);
    store_v(0);
    wait_async0();
    __syncthreads();

    for (int kb = 0; kb < nkb; kb++) {
        const int kbase = kb * 64;
        const int buf = kb & 1;
        if (kb + 1 < nkb) {              // overlap next block's fetch with compute
            async_k(kbase + 64, buf ^ 1);
            load_v(kbase + 64);
        }

        if (kbase <= qrow0 + 15) {       // wave has at least one live (q,key) pair
            // S = Q * K^T  (16 rows x 64 keys = 4 WMMA tiles, 2 k-steps)
            v8f s[4];
            #pragma unroll
            for (int nt = 0; nt < 4; nt++) s[nt] = (v8f){0.f,0.f,0.f,0.f,0.f,0.f,0.f,0.f};
            #pragma unroll
            for (int ks = 0; ks < 2; ks++)
                #pragma unroll
                for (int nt = 0; nt < 4; nt++) {
                    const bf16* p = &Ksm[buf][(nt * 16 + l16) * LDT + 32 * ks + 8 * kh];
                    v8bf lo = *(const v8bf*)p;
                    v8bf hi = *(const v8bf*)(p + 16);
                    v16bf kf;
                    #pragma unroll
                    for (int e = 0; e < 8; e++) { kf[e] = lo[e]; kf[e + 8] = hi[e]; }
                    s[nt] = __builtin_amdgcn_wmma_f32_16x16x32_bf16(
                        false, qf[ks], false, kf, (short)0, s[nt], false, false);
                }

            // scale (+ causal mask only on diagonal blocks) + online softmax
            const float scale = 0.125f;  // 1/sqrt(64)
            const bool diag = (kbase + 63 > qrow0);
            float mx[8];
            #pragma unroll
            for (int r = 0; r < 8; r++) {
                int qrow = qrow0 + r + 8 * kh;
                float m = -1e30f;
                #pragma unroll
                for (int nt = 0; nt < 4; nt++) {
                    float v = s[nt][r] * scale;
                    if (diag) {
                        int key = kbase + nt * 16 + l16;
                        v = (key > qrow) ? -1e30f : v;
                    }
                    s[nt][r] = v;
                    m = fmaxf(m, v);
                }
                mx[r] = m;
            }
            #pragma unroll
            for (int xm = 1; xm < 16; xm <<= 1)
                #pragma unroll
                for (int r = 0; r < 8; r++)
                    mx[r] = fmaxf(mx[r], __shfl_xor(mx[r], xm, 32));

            float fsc[8], rsum[8];
            #pragma unroll
            for (int r = 0; r < 8; r++) {
                float mnew = fmaxf(m_i[r], mx[r]);
                fsc[r] = __expf(m_i[r] - mnew);
                m_i[r] = mnew;
                float sum = 0.f;
                #pragma unroll
                for (int nt = 0; nt < 4; nt++) {
                    float p = __expf(s[nt][r] - mnew);
                    s[nt][r] = p;
                    sum += p;
                }
                rsum[r] = sum;
            }
            #pragma unroll
            for (int xm = 1; xm < 16; xm <<= 1)
                #pragma unroll
                for (int r = 0; r < 8; r++)
                    rsum[r] += __shfl_xor(rsum[r], xm, 32);
            #pragma unroll
            for (int r = 0; r < 8; r++) l_i[r] = l_i[r] * fsc[r] + rsum[r];
            #pragma unroll
            for (int nt = 0; nt < 4; nt++)
                #pragma unroll
                for (int r = 0; r < 8; r++)
                    o[nt][r] *= fsc[r];

            // C-layout -> A-layout via per-wave LDS round trip
            #pragma unroll
            for (int nt = 0; nt < 4; nt++)
                #pragma unroll
                for (int r = 0; r < 8; r++)
                    Pw[(r + 8 * kh) * LDT + nt * 16 + l16] = (bf16)s[nt][r];
            asm volatile("s_wait_dscnt 0" ::: "memory");

            // O += P * V
            #pragma unroll
            for (int ks = 0; ks < 2; ks++) {
                const bf16* pp = &Pw[l16 * LDT + 32 * ks + 8 * kh];
                v8bf lo = *(const v8bf*)pp;
                v8bf hi = *(const v8bf*)(pp + 16);
                v16bf pf;
                #pragma unroll
                for (int e = 0; e < 8; e++) { pf[e] = lo[e]; pf[e + 8] = hi[e]; }
                #pragma unroll
                for (int nt = 0; nt < 4; nt++) {
                    const bf16* vp = &Vt[buf][(nt * 16 + l16) * LDT + 32 * ks + 8 * kh];
                    v8bf vlo = *(const v8bf*)vp;
                    v8bf vhi = *(const v8bf*)(vp + 16);
                    v16bf vf;
                    #pragma unroll
                    for (int e = 0; e < 8; e++) { vf[e] = vlo[e]; vf[e + 8] = vhi[e]; }
                    o[nt] = __builtin_amdgcn_wmma_f32_16x16x32_bf16(
                        false, pf, false, vf, (short)0, o[nt], false, false);
                }
            }
        }

        if (kb + 1 < nkb) store_v((kb + 1) & 1);   // other buffer; prev reads fenced
        wait_async0();
        __syncthreads();
    }

    // Normalize and write bf16 output for the out-projection GEMM
    #pragma unroll
    for (int r = 0; r < 8; r++) {
        float inv = 1.f / l_i[r];
        size_t t = qrow0 + r + 8 * kh;
        bf16* op = attout + (bt + t) * D_MODEL + h * DH;
        #pragma unroll
        for (int nt = 0; nt < 4; nt++)
            op[nt * 16 + l16] = (bf16)(o[nt][r] * inv);
    }
}

// ---------------- host launch ----------------
extern "C" void kernel_launch(void* const* d_in, const int* in_sizes, int n_in,
                              void* d_out, int out_size, void* d_ws, size_t ws_size,
                              hipStream_t stream) {
    const float* x    = (const float*)d_in[0];
    const float* Wqkv = (const float*)d_in[1];
    const float* Wout = (const float*)d_in[2];

    bf16* x_bf    = (bf16*)d_ws;
    bf16* wqkv_bf = x_bf    + (size_t)M_TOTAL * D_MODEL;
    bf16* wout_bf = wqkv_bf + (size_t)3 * D_MODEL * D_MODEL;
    bf16* qkv_bf  = wout_bf + (size_t)D_MODEL * D_MODEL;
    bf16* att_bf  = qkv_bf  + (size_t)M_TOTAL * 3 * D_MODEL;

    const int n1 = M_TOTAL * D_MODEL;
    const int n2 = 3 * D_MODEL * D_MODEL;
    const int n3 = D_MODEL * D_MODEL;
    cvt_f32_bf16<<<(n1 / 4 + 255) / 256, 256, 0, stream>>>(x, x_bf, n1);
    cvt_f32_bf16<<<(n2 / 4 + 255) / 256, 256, 0, stream>>>(Wqkv, wqkv_bf, n2);
    cvt_f32_bf16<<<(n3 / 4 + 255) / 256, 256, 0, stream>>>(Wout, wout_bf, n3);

    // qkv = x @ Wqkv^T   [8192 x 3072]
    gemm_bf16<true><<<dim3(3 * D_MODEL / 128, M_TOTAL / 128), 256, 0, stream>>>(
        x_bf, wqkv_bf, qkv_bf, M_TOTAL, 3 * D_MODEL, D_MODEL);

    // flash attention  [4 x 16 x 2048 x 64]
    flash_attn<<<dim3(SEQ / 128, NH, BATCH), 256, 0, stream>>>(qkv_bf, att_bf);

    // out = att @ Wout^T  [8192 x 1024], fp32 output
    gemm_bf16<false><<<dim3(D_MODEL / 128, M_TOTAL / 128), 256, 0, stream>>>(
        att_bf, wout_bf, d_out, M_TOTAL, D_MODEL, D_MODEL);
}
